// SwinTransformerBlock_64012192579820
// MI455X (gfx1250) — compile-verified, matrix operand order checked by reading, not run on hardware
//
#include <hip/hip_runtime.h>
#include <hip/hip_bf16.h>
#include <cstdint>
#include <cstddef>

// ---------------------------------------------------------------------------
// Swin block, MI455X (gfx1250, wave32, WMMA 16x16x32 f16).
// Memory-bound workload: f16 activations everywhere, fused epilogues,
// async-to-LDS double-buffered GEMM staging (ASYNCcnt pipeline).
// ---------------------------------------------------------------------------

typedef _Float16 h16;
typedef __attribute__((ext_vector_type(2)))  _Float16 h16x2;
typedef __attribute__((ext_vector_type(16))) _Float16 v16h;
typedef __attribute__((ext_vector_type(8)))  float    v8f;

#define NHD   6
#define CDIM  192
#define HD    32
#define NTOK  49          // 7*7 tokens per window
#define HHW   112
#define NBAT  32
#define NWIN_B 256        // 16*16 windows per image
#define NWIN  (NBAT * NWIN_B)       // 8192
#define MROWS (NWIN * NTOK)         // 401408  (== NBAT * 112*112)
#define LSEQ  (HHW * HHW)           // 12544

// ---------------------------------------------------------------------------
// WMMA fragment loaders (A-operand layout per CDNA5 ISA 7.12.2; B assumed
// symmetric with N in place of M, matching W^T row-major == A row-major).
// Lane r = lane&15, half-K group = (lane>>4)*8; VGPR v holds K pair:
//   K = (v>=4 ? 16 : 0) + group + 2*(v&3)
// ---------------------------------------------------------------------------
__device__ inline v16h frag_ld_lds(const h16* p, int ld) {
  unsigned lane = threadIdx.x & 31u;
  unsigned r  = lane & 15u;
  unsigned kh = (lane >> 4) << 3;
  v16h f;
#pragma unroll
  for (int v = 0; v < 8; ++v) {
    unsigned k = ((v & 4) << 2) + kh + ((v & 3) << 1);
    h16x2 t = *(const h16x2*)(p + r * ld + k);
    f[2 * v] = t.x; f[2 * v + 1] = t.y;
  }
  return f;
}

// Global, row-major ld, rows >= nrows zero-filled (tile padding 49->64).
__device__ inline v16h frag_ld_g_rows(const h16* base, int ld, int nrows) {
  unsigned lane = threadIdx.x & 31u;
  int r = (int)(lane & 15u);
  unsigned kh = (lane >> 4) << 3;
  v16h f;
#pragma unroll
  for (int v = 0; v < 8; ++v) {
    unsigned k = ((v & 4) << 2) + kh + ((v & 3) << 1);
    if (r < nrows) {
      h16x2 t = *(const h16x2*)(base + (size_t)r * ld + k);
      f[2 * v] = t.x; f[2 * v + 1] = t.y;
    } else {
      f[2 * v] = (h16)0; f[2 * v + 1] = (h16)0;
    }
  }
  return f;
}

// V as B-operand for P@V: b[k][n] = V[k_token][dn + n]; k-padded to 64.
__device__ inline v16h frag_ld_vB(const h16* vbase, int k0, int dn) {
  unsigned lane = threadIdx.x & 31u;
  unsigned n  = lane & 15u;
  unsigned kh = (lane >> 4) << 3;
  v16h f;
#pragma unroll
  for (int v = 0; v < 8; ++v) {
    int k = k0 + (int)(((v & 4) << 2) + kh + ((v & 3) << 1));
    f[2 * v]     = (k     < NTOK) ? vbase[(size_t)k * HD + dn + n]       : (h16)0;
    f[2 * v + 1] = (k + 1 < NTOK) ? vbase[(size_t)(k + 1) * HD + dn + n] : (h16)0;
  }
  return f;
}

// ---------------------------------------------------------------------------
// f32 -> f16 weight conversion
// ---------------------------------------------------------------------------
__global__ void f32_to_f16(const float* __restrict__ src, h16* __restrict__ dst, int n) {
  int i = blockIdx.x * 256 + threadIdx.x;
  if (i < n) dst[i] = (h16)src[i];
}

// ---------------------------------------------------------------------------
// LayerNorm (C=192), one wave per row, fused destination mapping -> f16.
// ---------------------------------------------------------------------------
struct MapShiftWin {   // LN1: roll(-3,-3) + window partition scatter
  __device__ size_t operator()(int row, int c) const {
    int b = row / LSEQ, l = row - b * LSEQ;
    int h = l / HHW, w = l - h * HHW;
    int hr = h - 3; if (hr < 0) hr += HHW;
    int wr = w - 3; if (wr < 0) wr += HHW;
    int wy = hr / 7, r = hr - wy * 7;
    int wx = wr / 7, cc = wr - wx * 7;
    size_t wi = (size_t)b * NWIN_B + wy * 16 + wx;
    return (wi * NTOK + (r * 7 + cc)) * CDIM + c;
  }
};
struct MapIdent {      // LN2: natural layout
  __device__ size_t operator()(int row, int c) const {
    return (size_t)row * CDIM + c;
  }
};

template <class Map>
__global__ __launch_bounds__(256)
void ln_to_f16(const float* __restrict__ x, const float* __restrict__ g,
               const float* __restrict__ b, h16* __restrict__ out, Map map) {
  int row = blockIdx.x * 8 + (threadIdx.x >> 5);
  unsigned lane = threadIdx.x & 31u;
  const float* xr = x + (size_t)row * CDIM;
  float vv[6], s = 0.f;
#pragma unroll
  for (int i = 0; i < 6; ++i) { vv[i] = xr[lane + 32 * i]; s += vv[i]; }
#pragma unroll
  for (int off = 16; off; off >>= 1) s += __shfl_xor(s, off, 32);
  float mean = s * (1.f / 192.f);
  float var = 0.f;
#pragma unroll
  for (int i = 0; i < 6; ++i) { float d = vv[i] - mean; var += d * d; }
#pragma unroll
  for (int off = 16; off; off >>= 1) var += __shfl_xor(var, off, 32);
  float rinv = rsqrtf(var * (1.f / 192.f) + 1e-5f);
#pragma unroll
  for (int i = 0; i < 6; ++i) {
    int c = lane + 32 * i;
    out[map(row, c)] = (h16)((vv[i] - mean) * rinv * g[c] + b[c]);
  }
}

// ---------------------------------------------------------------------------
// Tiled WMMA GEMM:  D[m][n] = sum_k A[m][k] * W[n][k]   (out = A @ W^T)
// Block tile 128x64, BK=32, 256 threads = 8 wave32s; wave w owns 16 rows x 64
// cols (4 f32 accumulators). Staging uses GLOBAL_LOAD_ASYNC_TO_LDS_B128
// (CDNA5 async DMA, ASYNCcnt) with double-buffered LDS: tile i+1 streams in
// while tile i feeds the WMMAs; one barrier per K step.
// LDS stride 40 halves: 16B-aligned b128 targets, conflict-free h16x2 reads.
// ---------------------------------------------------------------------------
template <class Epi>
__global__ __launch_bounds__(256)
void gemm_epilogue(const h16* __restrict__ A, const h16* __restrict__ W,
                   int K, Epi epi) {
  constexpr int BM = 128, BN = 64, BK = 32, LD = BK + 8;   // 40 halves
  __shared__ __align__(16) h16 As[2][BM * LD];
  __shared__ __align__(16) h16 Bs[2][BN * LD];
  const int tid  = threadIdx.x;
  const int wave = tid >> 5;
  const int bm = blockIdx.y * BM;
  const int bn = blockIdx.x * BN;

  // Per-thread 16B staging chunks: A tile = 2 chunks, B tile = 1 chunk.
  const int ar0 = (tid * 8) >> 5,        ac0 = (tid * 8) & 31;
  const int ar1 = ((tid + 256) * 8) >> 5, ac1 = ((tid + 256) * 8) & 31;
  const int br  = (tid * 8) >> 5,        bc  = (tid * 8) & 31;
  const uint64_t Abase = (uint64_t)(uintptr_t)A;
  const uint64_t Wbase = (uint64_t)(uintptr_t)W;

  auto issue_tile = [&](int k0, int buf) {
    uint32_t lA0 = (uint32_t)(uintptr_t)&As[buf][ar0 * LD + ac0];
    uint32_t lA1 = (uint32_t)(uintptr_t)&As[buf][ar1 * LD + ac1];
    uint32_t lB  = (uint32_t)(uintptr_t)&Bs[buf][br  * LD + bc ];
    uint32_t gA0 = (uint32_t)(((bm + ar0) * K + k0 + ac0) * 2);
    uint32_t gA1 = (uint32_t)(((bm + ar1) * K + k0 + ac1) * 2);
    uint32_t gB  = (uint32_t)(((bn + br ) * K + k0 + bc ) * 2);
    asm volatile("global_load_async_to_lds_b128 %0, %1, %2"
                 :: "v"(lA0), "v"(gA0), "s"(Abase) : "memory");
    asm volatile("global_load_async_to_lds_b128 %0, %1, %2"
                 :: "v"(lA1), "v"(gA1), "s"(Abase) : "memory");
    asm volatile("global_load_async_to_lds_b128 %0, %1, %2"
                 :: "v"(lB),  "v"(gB),  "s"(Wbase) : "memory");
  };

  v8f acc[4];
#pragma unroll
  for (int i = 0; i < 4; ++i)
#pragma unroll
    for (int j = 0; j < 8; ++j) acc[i][j] = 0.f;

  const int nk = K / BK;
  issue_tile(0, 0);
  for (int i = 0; i < nk; ++i) {
    // Own async transfers for tile i have landed; barrier additionally proves
    // (a) all waves' tile-i data is in LDS and (b) all waves finished
    // computing tile i-1 from the buffer we are about to refill.
    asm volatile("s_wait_asynccnt 0x0" ::: "memory");
    __syncthreads();
    if (i + 1 < nk) issue_tile((i + 1) * BK, (i + 1) & 1);

    const int buf = i & 1;
    v16h afr = frag_ld_lds(&As[buf][wave * 16 * LD], LD);
#pragma unroll
    for (int nt = 0; nt < 4; ++nt) {
      v16h bfr = frag_ld_lds(&Bs[buf][nt * 16 * LD], LD);
      acc[nt] = __builtin_amdgcn_wmma_f32_16x16x32_f16(
          false, afr, false, bfr, (short)0, acc[nt], false, false);
    }
  }

  unsigned lane = tid & 31u;
  int rbase = bm + wave * 16 + (int)((lane >> 4) << 3);
  int cbase = bn + (int)(lane & 15u);
#pragma unroll
  for (int nt = 0; nt < 4; ++nt)
#pragma unroll
    for (int e = 0; e < 8; ++e)
      epi(rbase + e, cbase + nt * 16, acc[nt][e]);
}

// ---------------- GEMM epilogues ----------------
struct EpiQKV {      // split q/k/v, add bias, scale q by hd^-0.5, scatter per head
  const float* bias; h16* Q; h16* K; h16* V;
  __device__ void operator()(int m, int j, float a) const {
    float val = a + bias[j];
    int part = j / CDIM, rem = j - part * CDIM;
    int h = rem >> 5, d = rem & 31;
    if (part == 0) val *= 0.17677669529663687f;   // 32^-0.5
    h16* dst = (part == 0) ? Q : (part == 1 ? K : V);
    size_t wi = (size_t)(m / NTOK); int n = m - (int)wi * NTOK;
    dst[((wi * NHD + h) * NTOK + n) * HD + d] = (h16)val;
  }
};
struct EpiProj {     // + bias, window reverse, roll(+3,+3), residual add -> f32
  const float* bias; const float* x; float* x2;
  __device__ void operator()(int m, int c, float a) const {
    int wi = m / NTOK, n = m - wi * NTOK;
    int b = wi >> 8, wl = wi & 255;
    int wy = wl >> 4, wx = wl & 15;
    int r = n / 7, cc = n - r * 7;
    int h = wy * 7 + r + 3;  if (h >= HHW) h -= HHW;
    int w = wx * 7 + cc + 3; if (w >= HHW) w -= HHW;
    size_t idx = (((size_t)b * LSEQ) + h * HHW + w) * CDIM + c;
    x2[idx] = x[idx] + a + bias[c];
  }
};
struct EpiFC1 {      // + bias, exact GELU -> f16
  const float* bias; h16* out;
  __device__ void operator()(int m, int c, float a) const {
    float v = a + bias[c];
    float g = 0.5f * v * (1.f + erff(v * 0.70710678118654752f));
    out[(size_t)m * 768 + c] = (h16)g;
  }
};
struct EpiFC2 {      // + bias, residual add -> f32 output
  const float* bias; const float* x2; float* out;
  __device__ void operator()(int m, int c, float a) const {
    size_t i = (size_t)m * CDIM + c;
    out[i] = x2[i] + a + bias[c];
  }
};

// ---------------------------------------------------------------------------
// Windowed attention: one block (128 threads / 4 waves) per (window, head).
// 49x49 scores padded to 64x64; WMMA 16x16x32 f16 (K = hd = 32 exactly).
// Softmax denominator applied linearly to output rows (defer division).
// ---------------------------------------------------------------------------
__global__ __launch_bounds__(128)
void attn_kernel(const h16* __restrict__ Q, const h16* __restrict__ Kmat,
                 const h16* __restrict__ Vmat, const float* __restrict__ rpb,
                 h16* __restrict__ O) {
  const int wh   = blockIdx.x;
  const int wi   = wh / NHD;
  const int head = wh - wi * NHD;
  const h16* qb = Q    + (size_t)wh * NTOK * HD;
  const h16* kb = Kmat + (size_t)wh * NTOK * HD;
  const h16* vb = Vmat + (size_t)wh * NTOK * HD;

  __shared__ __align__(16) float S[64 * 66];
  __shared__ __align__(16) h16   P[64 * 66];
  __shared__ float rs[64];

  const int tid  = threadIdx.x;
  const int wave = tid >> 5;
  const unsigned lane = tid & 31u;

  // ---- S = q @ k^T (padded 64x64) ----
  {
    v16h qf = frag_ld_g_rows(qb + (size_t)(wave * 16) * HD, HD, NTOK - wave * 16);
    int rb = wave * 16 + (int)((lane >> 4) << 3);
    int cl = (int)(lane & 15u);
#pragma unroll
    for (int nt = 0; nt < 4; ++nt) {
      v16h kf = frag_ld_g_rows(kb + (size_t)(nt * 16) * HD, HD, NTOK - nt * 16);
      v8f s;
#pragma unroll
      for (int j = 0; j < 8; ++j) s[j] = 0.f;
      s = __builtin_amdgcn_wmma_f32_16x16x32_f16(
          false, qf, false, kf, (short)0, s, false, false);
#pragma unroll
      for (int e = 0; e < 8; ++e) S[(rb + e) * 66 + nt * 16 + cl] = s[e];
    }
  }
  __syncthreads();

  // ---- relative-position bias + SW-MSA mask + softmax (exp stored in P) ----
  if (tid < 64) {
    int r = tid;
    if (r < NTOK) {
      int r0 = r / 7, c0 = r - r0 * 7;
      int wl = wi & 255, wy = wl >> 4, wx = wl & 15;
      int hr = wy * 7 + r0, wc = wx * 7 + c0;           // rolled-frame coords
      int regq = (hr < 105 ? 0 : (hr < 109 ? 1 : 2)) * 3
               + (wc < 105 ? 0 : (wc < 109 ? 1 : 2));
      float mx = -1e30f;
      for (int m = 0; m < NTOK; ++m) {
        int r1 = m / 7, c1 = m - r1 * 7;
        int hk = wy * 7 + r1, wk = wx * 7 + c1;
        int regk = (hk < 105 ? 0 : (hk < 109 ? 1 : 2)) * 3
                 + (wk < 105 ? 0 : (wk < 109 ? 1 : 2));
        int bi = (r0 - r1 + 6) * 13 + (c0 - c1 + 6);
        float val = S[r * 66 + m] + rpb[bi * NHD + head]
                  + ((regq == regk) ? 0.f : -100.f);
        S[r * 66 + m] = val;
        mx = fmaxf(mx, val);
      }
      float sum = 0.f;
      for (int m = 0; m < 64; ++m) {
        float e = (m < NTOK) ? __expf(S[r * 66 + m] - mx) : 0.f;
        P[r * 66 + m] = (h16)e;
        sum += e;
      }
      rs[r] = sum;
    } else {
      for (int m = 0; m < 64; ++m) P[r * 66 + m] = (h16)0;
      rs[r] = 1.f;
    }
  }
  __syncthreads();

  // ---- O = P @ V  (wave = 16-row M tile, 2 N tiles of 16, K padded to 64) ----
  v8f oacc[2];
#pragma unroll
  for (int i = 0; i < 2; ++i)
#pragma unroll
    for (int j = 0; j < 8; ++j) oacc[i][j] = 0.f;

#pragma unroll
  for (int k0 = 0; k0 < 64; k0 += 32) {
    v16h pf = frag_ld_lds(P + wave * 16 * 66 + k0, 66);
#pragma unroll
    for (int nt = 0; nt < 2; ++nt) {
      v16h vf = frag_ld_vB(vb, k0, nt * 16);
      oacc[nt] = __builtin_amdgcn_wmma_f32_16x16x32_f16(
          false, pf, false, vf, (short)0, oacc[nt], false, false);
    }
  }
  {
    int rb = wave * 16 + (int)((lane >> 4) << 3);
    int cl = (int)(lane & 15u);
#pragma unroll
    for (int nt = 0; nt < 2; ++nt)
#pragma unroll
      for (int e = 0; e < 8; ++e) {
        int row = rb + e;
        if (row < NTOK) {
          float o = oacc[nt][e] / rs[row];
          O[((size_t)wi * NTOK + row) * CDIM + head * HD + nt * 16 + cl] = (h16)o;
        }
      }
  }
}

// ---------------------------------------------------------------------------
// Host-side orchestration. Workspace layout (MiB-aligned, phase overlapped):
//   [0..1)      f16 weights (~0.85 MB)
//   [1..296)    x2 (f32, 294 MB)        overlay: xw (f16, phase 1)
//   [296..444)  y16 (f16, 147 MB)       overlay: Q (f16, phase 1-2)
//   [444..1033) hmid (f16, 588 MB)      overlay: K, V, O (f16, phase 1-3)
// Peak ~1033 MiB.
// ---------------------------------------------------------------------------
extern "C" void kernel_launch(void* const* d_in, const int* in_sizes, int n_in,
                              void* d_out, int out_size, void* d_ws, size_t ws_size,
                              hipStream_t stream) {
  (void)in_sizes; (void)n_in; (void)out_size; (void)ws_size;

  const float* x      = (const float*)d_in[0];
  const float* n1g    = (const float*)d_in[1];
  const float* n1b    = (const float*)d_in[2];
  const float* qkv_w  = (const float*)d_in[3];
  const float* qkv_b  = (const float*)d_in[4];
  const float* rpb    = (const float*)d_in[5];
  const float* proj_w = (const float*)d_in[6];
  const float* proj_b = (const float*)d_in[7];
  const float* n2g    = (const float*)d_in[8];
  const float* n2b    = (const float*)d_in[9];
  const float* fc1_w  = (const float*)d_in[10];
  const float* fc1_b  = (const float*)d_in[11];
  const float* fc2_w  = (const float*)d_in[12];
  const float* fc2_b  = (const float*)d_in[13];
  float* out = (float*)d_out;

  char* ws = (char*)d_ws;
  const size_t MiB = 1ull << 20;
  h16* wq16 = (h16*)(ws);             // 576*192 = 110592
  h16* wp16 = wq16 + 110592;          // 192*192 = 36864
  h16* wf1  = wq16 + 147456;          // 768*192 = 147456
  h16* wf2  = wq16 + 294912;          // 192*768 = 147456
  float* x2 = (float*)(ws + 1 * MiB);
  h16* y16  = (h16*)(ws + 296 * MiB);
  h16* hmid = (h16*)(ws + 444 * MiB);
  // overlays (dead before their region's main tenant is written)
  h16* xw = (h16*)(ws + 1   * MiB);
  h16* Qb = (h16*)(ws + 296 * MiB);
  h16* Kb = (h16*)(ws + 444 * MiB);
  h16* Vb = (h16*)(ws + 592 * MiB);
  h16* Ob = (h16*)(ws + 740 * MiB);

  // Weight conversion f32 -> f16
  f32_to_f16<<<(110592 + 255) / 256, 256, 0, stream>>>(qkv_w, wq16, 110592);
  f32_to_f16<<<(36864  + 255) / 256, 256, 0, stream>>>(proj_w, wp16, 36864);
  f32_to_f16<<<(147456 + 255) / 256, 256, 0, stream>>>(fc1_w, wf1, 147456);
  f32_to_f16<<<(147456 + 255) / 256, 256, 0, stream>>>(fc2_w, wf2, 147456);

  // LN1 + shift + window partition -> xw (f16)
  ln_to_f16<<<MROWS / 8, 256, 0, stream>>>(x, n1g, n1b, xw, MapShiftWin{});

  // QKV GEMM (M=401408, N=576, K=192)
  gemm_epilogue<<<dim3(576 / 64, MROWS / 128), 256, 0, stream>>>(
      xw, wq16, 192, EpiQKV{qkv_b, Qb, Kb, Vb});

  // Windowed attention (8192 windows x 6 heads)
  attn_kernel<<<NWIN * NHD, 128, 0, stream>>>(Qb, Kb, Vb, rpb, Ob);

  // Projection GEMM + window reverse + roll back + residual -> x2 (f32)
  gemm_epilogue<<<dim3(192 / 64, MROWS / 128), 256, 0, stream>>>(
      Ob, wp16, 192, EpiProj{proj_b, x, x2});

  // LN2 -> y16 (f16)
  ln_to_f16<<<MROWS / 8, 256, 0, stream>>>(x2, n2g, n2b, y16, MapIdent{});

  // FC1 GEMM + GELU (M=401408, N=768, K=192)
  gemm_epilogue<<<dim3(768 / 64, MROWS / 128), 256, 0, stream>>>(
      y16, wf1, 192, EpiFC1{fc1_b, hmid});

  // FC2 GEMM + residual -> out (M=401408, N=192, K=768)
  gemm_epilogue<<<dim3(192 / 64, MROWS / 128), 256, 0, stream>>>(
      hmid, wf2, 768, EpiFC2{fc2_b, x2, out});
}
